// PointEncoderV5_87995289960542
// MI455X (gfx1250) — matrix operand order repaired and checked by reading, not compiled, not run on hardware
//
#include <hip/hip_runtime.h>

#define NN 5
#define AA 900
#define ADIM 8
#define DD 256
#define HH 8
#define DH 32
#define KTOPC 300
#define MV 1200   // valid m
#define MP 1216   // padded m (38*32)
#define AP 912    // padded anchors (57*16)
#define NTA 57

typedef __attribute__((ext_vector_type(16))) __bf16 v16bf;
typedef __attribute__((ext_vector_type(8))) float v8f;
typedef __attribute__((ext_vector_type(4))) unsigned int u32x4;
typedef __attribute__((ext_vector_type(8))) unsigned int u32x8;

union BFrag { v16bf v; unsigned int u[8]; unsigned short s[16]; };

__device__ __forceinline__ unsigned short f2bf(float x) {
  unsigned int u = __float_as_uint(x);
  return (unsigned short)((u + 0x7FFFu + ((u >> 16) & 1u)) >> 16);
}
__device__ __forceinline__ float bf2f(unsigned short s) {
  return __uint_as_float(((unsigned int)s) << 16);
}
__device__ __forceinline__ v8f wmma_bf16(v16bf a, v16bf b, v8f c) {
  return __builtin_amdgcn_wmma_f32_16x16x32_bf16(false, a, false, b, (short)0, c, false, false);
}

// A fragment: 16x32 bf16, row-major source, row = lane&15.
// VGPR v holds K = (v%4)*2 + (v/4)*16 + (lane>>4)*8  (+1)
__device__ __forceinline__ void load_a_frag(BFrag& f, const unsigned short* A, int row,
                                            int kbase, int ld, int lane) {
  const unsigned short* p = A + (size_t)row * ld + kbase + ((lane >> 4) * 8);
#pragma unroll
  for (int v = 0; v < 4; ++v) {
    f.u[v]     = *(const unsigned int*)(p + 2 * v);
    f.u[4 + v] = *(const unsigned int*)(p + 16 + 2 * v);
  }
}
// B fragment: 32x16 bf16, col = lane&15, source stored K-contiguous per column:
// element (k,col) at BT[col*ldk + k].  VGPR v holds K = (v%4)*2 + (v/4)*16 + (lane>>4)*8 (+1)
__device__ __forceinline__ void load_b_frag(BFrag& f, const unsigned short* BT, int col,
                                            int kbase, int ldk, int lane) {
  const unsigned short* p = BT + (size_t)col * ldk + kbase + ((lane >> 4) * 8);
#pragma unroll
  for (int v = 0; v < 4; ++v) {
    f.u[v]     = *(const unsigned int*)(p + 2 * v);
    f.u[4 + v] = *(const unsigned int*)(p + 16 + 2 * v);
  }
}

// ---------------- weight prep: bf16 transposed copies ----------------
__global__ void k_prep_w(const float* wq, const float* wk, const float* wv, const float* wo,
                         unsigned short* wqT, unsigned short* wkT, unsigned short* wvT,
                         unsigned short* woT) {
  int gid = blockIdx.x * blockDim.x + threadIdx.x;
  if (gid >= 4 * DD * DD) return;
  int wsel = gid / (DD * DD);
  int r = gid % (DD * DD);
  int c = r >> 8, k = r & 255;
  const float* s = wsel == 0 ? wq : wsel == 1 ? wk : wsel == 2 ? wv : wo;
  unsigned short* d = wsel == 0 ? wqT : wsel == 1 ? wkT : wsel == 2 ? wvT : woT;
  d[c * DD + k] = f2bf(s[k * DD + c]);
}

// ---------------- roi score + stable top-k via bitonic sort ----------------
__global__ void __launch_bounds__(1024) k_roi_topk(const float* bfeat, const float* wroi,
                                                   const float* broi, int* top_idx, float* conf) {
  __shared__ unsigned long long keys[1024];
  int n = blockIdx.x, t = threadIdx.x;
  unsigned long long key = 0ull;
  if (t < AA) {
    float acc = broi[0];
    const float* row = bfeat + ((size_t)n * AA + t) * DD;
    for (int d = 0; d < DD; ++d) acc += row[d] * wroi[d];
    float sc = 1.f / (1.f + __expf(-acc));
    unsigned int u = __float_as_uint(sc);
    u = (u & 0x80000000u) ? ~u : (u | 0x80000000u);  // order-preserving map
    key = ((unsigned long long)u << 32) | (unsigned int)(~t);
  }
  keys[t] = key;
  __syncthreads();
  for (int k = 2; k <= 1024; k <<= 1) {
    for (int j = k >> 1; j > 0; j >>= 1) {
      int p = t ^ j;
      if (p > t) {
        bool desc = ((t & k) == 0);
        unsigned long long a = keys[t], b = keys[p];
        bool sw = desc ? (a < b) : (a > b);
        if (sw) { keys[t] = b; keys[p] = a; }
      }
      __syncthreads();
    }
  }
  if (t < KTOPC) {
    unsigned long long k2 = keys[t];
    int idx = (int)(~(unsigned int)k2);
    unsigned int hi = (unsigned int)(k2 >> 32);
    unsigned int u = (hi & 0x80000000u) ? (hi & 0x7FFFFFFFu) : ~hi;
    float s = __uint_as_float(u);
    top_idx[n * KTOPC + t] = idx;
    conf[n * KTOPC + t] = (s > 0.5f) ? 1.f : 0.f;
  }
}

// ---------------- gathers ----------------
__global__ void k_gather(const float* bfeat, const float* i2ja, const int* top_idx,
                         const float* conf, unsigned short* keyf, unsigned short* kmaskT,
                         float* i2j) {
  int m = blockIdx.x, n = blockIdx.y, d = threadIdx.x;
  float kf = 0.f, cm = 0.f;
  int idx = 0;
  bool valid = m < MV;
  int jj = m / KTOPC, kk = m % KTOPC;
  int j = jj + (jj >= n);
  if (valid) {
    idx = top_idx[j * KTOPC + kk];
    cm = conf[j * KTOPC + kk];
    kf = bfeat[((size_t)j * AA + idx) * DD + d];
  }
  keyf[((size_t)n * MP + m) * DD + d] = f2bf(kf);
  kmaskT[((size_t)n * DD + d) * MP + m] = f2bf(kf * cm);
  if (d < ADIM) {
    float val = valid ? i2ja[(((size_t)n * NN + j) * AA + idx) * ADIM + d] : 0.f;
    i2j[((size_t)n * MP + m) * ADIM + d] = val;
  }
}

// ---------------- transform MLP (20x12 BN -> fc1 -> fc2) ----------------
__global__ void __launch_bounds__(256) k_tmlp(const float* tmat, const float* bn_g,
                                              const float* bn_b, const float* fc1w,
                                              const float* fc1b, const float* fc2w,
                                              const float* fc2b, float* tbuf) {
  __shared__ float xbn[20 * 12];
  __shared__ float hbuf[20 * DD];
  __shared__ float mu[12], iv[12];
  int t = threadIdx.x;
  if (t < 240) {
    int r = t / 12, c = t % 12;
    int i = r / 4, jj = r % 4;
    int j = jj + (jj >= i);
    int rr = c / 4, cc = c % 4;
    xbn[t] = tmat[(((size_t)i * NN + j) * 4 + rr) * 4 + cc];
  }
  __syncthreads();
  if (t < 12) {
    float s = 0.f, s2 = 0.f;
    for (int r = 0; r < 20; ++r) { float x = xbn[r * 12 + t]; s += x; s2 += x * x; }
    float m = s * (1.f / 20.f);
    float v = s2 * (1.f / 20.f) - m * m;
    mu[t] = m; iv[t] = rsqrtf(v + 1e-5f);
  }
  __syncthreads();
  if (t < 240) {
    int c = t % 12;
    xbn[t] = (xbn[t] - mu[c]) * iv[c] * bn_g[c] + bn_b[c];
  }
  __syncthreads();
  for (int r = 0; r < 20; ++r) {
    float acc = fc1b[t];
    for (int e = 0; e < 12; ++e) acc += xbn[r * 12 + e] * fc1w[e * DD + t];
    hbuf[r * DD + t] = fmaxf(acc, 0.f);
  }
  __syncthreads();
  for (int r = 0; r < 20; ++r) {
    float acc = fc2b[t];
    for (int e = 0; e < DD; ++e) acc += hbuf[r * DD + e] * fc2w[e * DD + t];
    tbuf[r * DD + t] = acc;
  }
}

// ---------------- k_in = key_feat + i2j@anc_w + anc_b + trans_enc ----------------
__global__ void k_keypos(const unsigned short* keyf, const float* i2j, const float* ancw,
                         const float* ancb, const float* tbuf, unsigned short* kin) {
  int m = blockIdx.x, n = blockIdx.y, d = threadIdx.x;
  float acc = 0.f;
  if (m < MV) {
    acc = bf2f(keyf[((size_t)n * MP + m) * DD + d]);
    const float* ir = i2j + ((size_t)n * MP + m) * ADIM;
#pragma unroll
    for (int e = 0; e < ADIM; ++e) acc += ir[e] * ancw[e * DD + d];
    acc += ancb[d] + tbuf[((size_t)n * 4 + m / KTOPC) * DD + d];
  }
  kin[((size_t)n * MP + m) * DD + d] = f2bf(acc);
}

// ---------------- agg GEMM: inside(900x1200) @ masked_feat(1200x256), A built on the fly
__global__ void __launch_bounds__(256) k_agg(const float* b_anchor, const float* i2j,
                                             const unsigned short* kmaskT, const float* bfeat,
                                             const float* aemb, float* bfu,
                                             unsigned short* qbf) {
  int n = blockIdx.y;
  int wid = threadIdx.x >> 5, lane = threadIdx.x & 31;
  int tile = blockIdx.x * 8 + wid;
  int tm = tile >> 4, tn = tile & 15;
  int hf = lane >> 4, ln16 = lane & 15;
  int arow = tm * 16 + ln16;
  float mnx = 0, mny = 0, mnz = 0, mxx = 0, mxy = 0, mxz = 0;
  bool valid = arow < AA;
  if (valid) {
    const float* an = b_anchor + ((size_t)n * AA + arow) * ADIM;
    float x = an[0], y = an[1], z = an[2];
    float w = __expf(an[3]), l = __expf(an[4]), h = __expf(an[5]);
    float sy = an[6], cy = an[7];
    float r2 = sy * sy + cy * cy;
    float ca, sa;
    if (r2 > 0.f) { float inv = rsqrtf(r2); ca = fabsf(cy) * inv; sa = fabsf(sy) * inv; }
    else { ca = 1.f; sa = 0.f; }
    float ex = 0.5f * (l * ca + w * sa);
    float ey = 0.5f * (l * sa + w * ca);
    float ez = 0.5f * h;
    mnx = x - ex; mxx = x + ex; mny = y - ey; mxy = y + ey; mnz = z - ez; mxz = z + ez;
  }
  const float* cen = i2j + (size_t)n * MP * ADIM;
  const unsigned short* Bn = kmaskT + (size_t)n * DD * MP;
  v8f acc = {};
  for (int kc = 0; kc < MP / 32; ++kc) {
    if (kc + 1 < MP / 32) __builtin_prefetch(cen + (size_t)(kc + 1) * 32 * ADIM, 0, 0);
    BFrag fa, fb;
#pragma unroll
    for (int v = 0; v < 8; ++v) {
      int kb = (v & 3) * 2 + (v >> 2) * 16 + hf * 8;
#pragma unroll
      for (int e = 0; e < 2; ++e) {
        int m = kc * 32 + kb + e;
        const float* c3 = cen + (size_t)m * ADIM;
        float cx = c3[0], cyy = c3[1], cz = c3[2];
        bool in = valid && cx >= mnx && cx <= mxx && cyy >= mny && cyy <= mxy &&
                  cz >= mnz && cz <= mxz;
        fa.s[2 * v + e] = in ? (unsigned short)0x3F80 : (unsigned short)0;
      }
    }
    load_b_frag(fb, Bn, tn * 16 + ln16, kc * 32, MP, lane);
    acc = wmma_bf16(fa.v, fb.v, acc);
  }
  int col = tn * 16 + ln16;
#pragma unroll
  for (int v = 0; v < 8; ++v) {
    int row = tm * 16 + v + hf * 8;
    size_t po = ((size_t)n * AP + row) * DD + col;
    float bv = 0.f, qv = 0.f;
    if (row < AA) {
      size_t pi = ((size_t)n * AA + row) * DD + col;
      bv = bfeat[pi] + acc[v];
      qv = bv + aemb[pi];
    }
    bfu[po] = bv;
    qbf[po] = f2bf(qv);
  }
}

// ---------------- lm = bfu @ lam_w + lam_b (per head) ----------------
__global__ void k_lm(const float* bfu, const float* lamw, const float* lamb, float* lm) {
  int n = blockIdx.x, a = threadIdx.x;
  if (a >= AA) return;
  const float* row = bfu + ((size_t)n * AP + a) * DD;
  for (int h = 0; h < HH; ++h) {
    float acc = lamb[h];
    for (int d = 0; d < DD; ++d) acc += row[d] * lamw[d * HH + h];
    lm[((size_t)n * HH + h) * AP + a] = acc;
  }
}

// ---------------- generic WMMA GEMM with TDM + async-LDS staging ----------------
// C = A(bf16, Mpad x 256) @ W(256x256) + bias
// A tile (16x256, shared by all 8 waves) staged via global_load_async_to_lds_b128.
// Each wave's 16 weight columns staged by the Tensor Data Mover (2 halves of K=128).
// mode 0: bf16 row-major out; mode 1: bf16 transposed out [col][row]; mode 2: f32 out
__global__ void __launch_bounds__(256) k_gemm(const unsigned short* A, const unsigned short* WT,
                                              const float* bias, void* out, int Mpad, int mode) {
  __shared__ unsigned short ldsA[16 * DD];        // 8 KB: A rows tm*16..+15
  __shared__ unsigned short ldsW[8 * 16 * 128];   // 32 KB: per-wave 16 cols x 128 K
  int n = blockIdx.y;
  int wid = threadIdx.x >> 5, lane = threadIdx.x & 31;
  int tile = blockIdx.x * 8 + wid;
  int tm = tile >> 4, tn = tile & 15;
  int ln16 = lane & 15, hf = lane >> 4;
  int colbase = (blockIdx.x & 1) * 128;  // waves of this block cover cols colbase..+127
  const unsigned short* An = A + (size_t)n * Mpad * DD;

  // ---- async stage of shared A tile (512 x 16B chunks across 256 threads) ----
  {
    unsigned long long gbase = (unsigned long long)(An + (size_t)(tm * 16) * DD);
    unsigned lbase = (unsigned)(size_t)(void*)ldsA;
#pragma unroll
    for (int it = 0; it < 2; ++it) {
      int idx = it * 256 + (int)threadIdx.x;
      unsigned long long ga = gbase + (unsigned long long)idx * 16ull;
      unsigned la = lbase + (unsigned)idx * 16u;
      asm volatile("global_load_async_to_lds_b128 %0, %1, off" :: "v"(la), "v"(ga) : "memory");
    }
  }
  asm volatile("s_wait_asynccnt 0x0" ::: "memory");
  __syncthreads();

  v8f acc = {};
  const unsigned short* wlds = ldsW + (size_t)wid * 16 * 128;
  unsigned wldsaddr = (unsigned)(size_t)(const void*)wlds;
  for (int hk = 0; hk < 2; ++hk) {
    // ---- TDM: DMA this wave's 16 weight columns (K half hk) into LDS ----
    {
      unsigned long long g =
          (unsigned long long)(WT + ((size_t)(colbase + wid * 16)) * DD + hk * 128);
      u32x4 g0;
      g0.x = 1u;                                            // count=1 valid descriptor
      g0.y = wldsaddr;                                      // lds_addr
      g0.z = (unsigned)g;                                   // global_addr[31:0]
      g0.w = ((unsigned)(g >> 32) & 0x01FFFFFFu) | (2u << 30);  // addr[56:32] | type=2
      u32x8 g1;
      g1[0] = 0x00010000u;          // workgroup_mask=0, data_size=1 (2 bytes)
      g1[1] = (128u & 0xFFFFu) << 16;            // tensor_dim0 = 128 (lo16)
      g1[2] = 0u | (16u << 16);                  // tensor_dim0 hi16 | tensor_dim1=16 lo16
      g1[3] = 0u | (128u << 16);                 // tensor_dim1 hi16 | tile_dim0=128
      g1[4] = 16u;                               // tile_dim1=16, tile_dim2=0
      g1[5] = 256u;                              // tensor_dim0_stride = 256 (lo32)
      g1[6] = 0u;                                // stride0 hi16 | stride1 lo16
      g1[7] = 0u;                                // stride1 hi32
      asm volatile("tensor_load_to_lds %0, %1" :: "s"(g0), "s"(g1) : "memory");
    }
    asm volatile("s_wait_tensorcnt 0x0" ::: "memory");
#pragma unroll
    for (int kc4 = 0; kc4 < 4; ++kc4) {
      int kc = hk * 4 + kc4;
      BFrag fa, fb;
      load_a_frag(fa, ldsA, ln16, kc * 32, DD, lane);
      load_b_frag(fb, wlds, ln16, kc4 * 32, 128, lane);
      acc = wmma_bf16(fa.v, fb.v, acc);
    }
    // ensure LDS reads retired before the next TDM overwrites this wave's region
    asm volatile("s_wait_dscnt 0x0" ::: "memory");
  }

  int col = tn * 16 + ln16;
  float bv = bias[col];
#pragma unroll
  for (int v = 0; v < 8; ++v) {
    int r = tm * 16 + v + hf * 8;
    float val = acc[v] + bv;
    if (mode == 0) {
      ((unsigned short*)out)[((size_t)n * Mpad + r) * DD + col] = f2bf(val);
    } else if (mode == 1) {
      ((unsigned short*)out)[((size_t)n * DD + col) * Mpad + r] = f2bf(val);
    } else {
      ((float*)out)[((size_t)n * Mpad + r) * DD + col] = val;
    }
  }
}

// ---------------- fused flash attention: one wave per (n, h, 16-anchor tile) ----------------
__global__ void __launch_bounds__(32) k_attn(const unsigned short* qh, const unsigned short* kh,
                                             const unsigned short* vhT, const float* lm,
                                             const float* b_anchor, const float* i2j,
                                             unsigned short* ao) {
  __shared__ unsigned short Pbuf[16 * 32];
  int ta = blockIdx.x, h = blockIdx.y, n = blockIdx.z;
  int lane = threadIdx.x;
  int ln16 = lane & 15, hf = lane >> 4;
  const unsigned short* qn = qh + (size_t)n * AP * DD;
  const unsigned short* kn = kh + (size_t)n * MP * DD;
  const unsigned short* vn = vhT + (size_t)n * DD * MP;
  BFrag fq;
  load_a_frag(fq, qn, ta * 16 + ln16, h * DH, DD, lane);
  float ax[8], ay[8], lmv[8], rmax[8], rsum[8];
#pragma unroll
  for (int v = 0; v < 8; ++v) {
    int r = ta * 16 + v + hf * 8;
    if (r < AA) {
      const float* an = b_anchor + ((size_t)n * AA + r) * ADIM;
      ax[v] = an[0]; ay[v] = an[1];
      lmv[v] = lm[((size_t)n * HH + h) * AP + r];
    } else { ax[v] = 0.f; ay[v] = 0.f; lmv[v] = 0.f; }
    rmax[v] = -3.0e38f; rsum[v] = 0.f;
  }
  const float* cen = i2j + (size_t)n * MP * ADIM;
  v8f O0 = {}, O1 = {};
  const float scale = 0.1767766952966369f;  // 1/sqrt(32)
  for (int mc = 0; mc < MP / 32; ++mc) {
    int m0 = mc * 32;
    BFrag fb0, fb1;
    load_b_frag(fb0, kn, m0 + ln16, h * DH, DD, lane);
    load_b_frag(fb1, kn, m0 + 16 + ln16, h * DH, DD, lane);
    v8f S0 = {}, S1 = {};
    S0 = wmma_bf16(fq.v, fb0.v, S0);
    S1 = wmma_bf16(fq.v, fb1.v, S1);
    int c0 = m0 + ln16, c1 = c0 + 16;
    float cx0 = cen[(size_t)c0 * ADIM], cy0 = cen[(size_t)c0 * ADIM + 1];
    float cx1 = cen[(size_t)c1 * ADIM], cy1 = cen[(size_t)c1 * ADIM + 1];
    bool ok0 = c0 < MV, ok1 = c1 < MV;
    float p0a[8], p1a[8];
#pragma unroll
    for (int v = 0; v < 8; ++v) {
      float dx0 = ax[v] - cx0, dy0 = ay[v] - cy0;
      float dx1 = ax[v] - cx1, dy1 = ay[v] - cy1;
      float d0 = sqrtf(dx0 * dx0 + dy0 * dy0 + 1e-12f);
      float d1 = sqrtf(dx1 * dx1 + dy1 * dy1 + 1e-12f);
      float s0 = ok0 ? (S0[v] * scale - log1pf(d0) * lmv[v]) : -3.0e38f;
      float s1 = ok1 ? (S1[v] * scale - log1pf(d1) * lmv[v]) : -3.0e38f;
      float tmax = fmaxf(s0, s1);
#pragma unroll
      for (int off = 1; off < 16; off <<= 1) tmax = fmaxf(tmax, __shfl_xor(tmax, off, 32));
      float nm = fmaxf(rmax[v], tmax);
      float corr = __expf(rmax[v] - nm);
      float p0 = ok0 ? __expf(s0 - nm) : 0.f;
      float p1 = ok1 ? __expf(s1 - nm) : 0.f;
      float rs = p0 + p1;
#pragma unroll
      for (int off = 1; off < 16; off <<= 1) rs += __shfl_xor(rs, off, 32);
      rsum[v] = rsum[v] * corr + rs;
      rmax[v] = nm;
      O0[v] *= corr; O1[v] *= corr;
      p0a[v] = p0; p1a[v] = p1;
    }
    __syncthreads();
#pragma unroll
    for (int v = 0; v < 8; ++v) {
      int r = v + hf * 8;
      Pbuf[r * 32 + ln16] = f2bf(p0a[v]);
      Pbuf[r * 32 + 16 + ln16] = f2bf(p1a[v]);
    }
    __syncthreads();
    BFrag fp;
    {
      const unsigned short* p = Pbuf + ln16 * 32 + hf * 8;
#pragma unroll
      for (int v = 0; v < 4; ++v) {
        fp.u[v]     = *(const unsigned int*)(p + 2 * v);
        fp.u[4 + v] = *(const unsigned int*)(p + 16 + 2 * v);
      }
    }
    BFrag fv0, fv1;
    load_b_frag(fv0, vn, h * DH + ln16, m0, MP, lane);
    load_b_frag(fv1, vn, h * DH + 16 + ln16, m0, MP, lane);
    O0 = wmma_bf16(fp.v, fv0.v, O0);
    O1 = wmma_bf16(fp.v, fv1.v, O1);
  }
#pragma unroll
  for (int v = 0; v < 8; ++v) {
    int r = ta * 16 + v + hf * 8;
    float inv = 1.f / rsum[v];
    size_t base = ((size_t)n * AP + r) * DD + h * DH;
    ao[base + ln16] = f2bf(O0[v] * inv);
    ao[base + 16 + ln16] = f2bf(O1[v] * inv);
  }
}

// ---------------- residual + LayerNorm ----------------
__global__ void __launch_bounds__(256) k_ln(const float* bfu, const float* oout, const float* g,
                                            const float* b, float* out) {
  __shared__ float s1[DD], s2[DD];
  int row = blockIdx.x;
  int n = row / AA, a = row % AA;
  int d = threadIdx.x;
  size_t pi = ((size_t)n * AP + a) * DD + d;
  float x = bfu[pi] + oout[pi];
  s1[d] = x; s2[d] = x * x;
  __syncthreads();
  for (int st = 128; st > 0; st >>= 1) {
    if (d < st) { s1[d] += s1[d + st]; s2[d] += s2[d + st]; }
    __syncthreads();
  }
  float mean = s1[0] * (1.f / DD);
  float var = s2[0] * (1.f / DD) - mean * mean;
  out[((size_t)n * AA + a) * DD + d] = (x - mean) * rsqrtf(var + 1e-5f) * g[d] + b[d];
}

extern "C" void kernel_launch(void* const* d_in, const int* in_sizes, int n_in, void* d_out,
                              int out_size, void* d_ws, size_t ws_size, hipStream_t stream) {
  (void)in_sizes; (void)n_in; (void)out_size; (void)ws_size;
  const float* i2ja = (const float*)d_in[0];
  const float* b_anchor = (const float*)d_in[1];
  const float* bfeat = (const float*)d_in[2];
  const float* tmat = (const float*)d_in[3];
  const float* aemb = (const float*)d_in[4];
  const float* wroi = (const float*)d_in[5];
  const float* broi = (const float*)d_in[6];
  const float* bng = (const float*)d_in[7];
  const float* bnb = (const float*)d_in[8];
  const float* fc1w = (const float*)d_in[9];
  const float* fc1b = (const float*)d_in[10];
  const float* fc2w = (const float*)d_in[11];
  const float* fc2b = (const float*)d_in[12];
  const float* ancw = (const float*)d_in[13];
  const float* ancb = (const float*)d_in[14];
  const float* lamw = (const float*)d_in[15];
  const float* lamb = (const float*)d_in[16];
  const float* wq = (const float*)d_in[17];
  const float* bq = (const float*)d_in[18];
  const float* wk = (const float*)d_in[19];
  const float* bk = (const float*)d_in[20];
  const float* wv = (const float*)d_in[21];
  const float* bv = (const float*)d_in[22];
  const float* wo = (const float*)d_in[23];
  const float* bo = (const float*)d_in[24];
  const float* lng = (const float*)d_in[25];
  const float* lnb = (const float*)d_in[26];
  float* out = (float*)d_out;

  char* base = (char*)d_ws;
  size_t off = 0;
  auto alloc = [&](size_t bytes) -> char* {
    char* p = base + off;
    off = (off + bytes + 255) & ~(size_t)255;
    return p;
  };
  int* top_idx = (int*)alloc((size_t)NN * KTOPC * 4);
  float* conf = (float*)alloc((size_t)NN * KTOPC * 4);
  unsigned short* keyf = (unsigned short*)alloc((size_t)NN * MP * DD * 2);
  unsigned short* kmaskT = (unsigned short*)alloc((size_t)NN * DD * MP * 2);
  float* i2j = (float*)alloc((size_t)NN * MP * ADIM * 4);
  float* tbuf = (float*)alloc((size_t)20 * DD * 4);
  unsigned short* kin = (unsigned short*)alloc((size_t)NN * MP * DD * 2);
  float* bfu = (float*)alloc((size_t)NN * AP * DD * 4);
  unsigned short* qbf = (unsigned short*)alloc((size_t)NN * AP * DD * 2);
  float* lmbuf = (float*)alloc((size_t)NN * HH * AP * 4);
  unsigned short* wqT = (unsigned short*)alloc((size_t)DD * DD * 2);
  unsigned short* wkT = (unsigned short*)alloc((size_t)DD * DD * 2);
  unsigned short* wvT = (unsigned short*)alloc((size_t)DD * DD * 2);
  unsigned short* woT = (unsigned short*)alloc((size_t)DD * DD * 2);
  unsigned short* qhb = (unsigned short*)alloc((size_t)NN * AP * DD * 2);
  unsigned short* khb = (unsigned short*)alloc((size_t)NN * MP * DD * 2);
  unsigned short* vhT = (unsigned short*)alloc((size_t)NN * DD * MP * 2);
  unsigned short* ao = (unsigned short*)alloc((size_t)NN * AP * DD * 2);
  float* oout = (float*)alloc((size_t)NN * AP * DD * 4);

  k_prep_w<<<(4 * DD * DD + 255) / 256, 256, 0, stream>>>(wq, wk, wv, wo, wqT, wkT, wvT, woT);
  k_roi_topk<<<NN, 1024, 0, stream>>>(bfeat, wroi, broi, top_idx, conf);
  k_gather<<<dim3(MP, NN), DD, 0, stream>>>(bfeat, i2ja, top_idx, conf, keyf, kmaskT, i2j);
  k_tmlp<<<1, DD, 0, stream>>>(tmat, bng, bnb, fc1w, fc1b, fc2w, fc2b, tbuf);
  k_keypos<<<dim3(MP, NN), DD, 0, stream>>>(keyf, i2j, ancw, ancb, tbuf, kin);
  k_agg<<<dim3(114, NN), 256, 0, stream>>>(b_anchor, i2j, kmaskT, bfeat, aemb, bfu, qbf);
  k_lm<<<NN, AA, 0, stream>>>(bfu, lamw, lamb, lmbuf);
  k_gemm<<<dim3(114, NN), 256, 0, stream>>>(qbf, wqT, bq, (void*)qhb, AP, 0);
  k_gemm<<<dim3(152, NN), 256, 0, stream>>>(kin, wkT, bk, (void*)khb, MP, 0);
  k_gemm<<<dim3(152, NN), 256, 0, stream>>>(keyf, wvT, bv, (void*)vhT, MP, 1);
  k_attn<<<dim3(NTA, HH, NN), 32, 0, stream>>>(qhb, khb, vhT, lmbuf, b_anchor, i2j, ao);
  k_gemm<<<dim3(114, NN), 256, 0, stream>>>(ao, woT, bo, (void*)oout, AP, 2);
  k_ln<<<NN * AA, DD, 0, stream>>>(bfu, oout, lng, lnb, out);
}
